// GRAN_68599217652301
// MI455X (gfx1250) — compile-verified, matrix operand order checked by reading, not run on hardware
//
#include <hip/hip_runtime.h>
#include <hip/hip_bf16.h>

// MI455X / gfx1250: wave32, WMMA 16x16x32 bf16->f32.
// ~425 GFLOP of GEMMs vs ~1.5 GB of HBM traffic (23.3 TB/s) -> compute-bound,
// so every GEMM runs on v_wmma_f32_16x16x32_bf16 with fp32 accumulation.
// Tiles stream global->LDS via GLOBAL_LOAD_ASYNC_TO_LDS_B128 (ASYNCcnt),
// double-buffered: one s_wait_asynccnt + one barrier per K-tile.
// Weights are pre-transposed at staging so all weight GEMMs use the
// row-major (BT) tile path; only attn@v's v-tile needs an in-LDS transpose.

typedef __attribute__((ext_vector_type(16))) __bf16 v16bf;
typedef __attribute__((ext_vector_type(8)))  __bf16 v8bf;
typedef __attribute__((ext_vector_type(8)))  float  v8f;

#define BMT 128
#define BNT 128
#define BKT 32
#define LDSS 40   // bf16 elems per LDS row = 80 bytes (keeps 16B chunk alignment)

enum { EPI_ELU=0, EPI_DUAL=1, EPI_SILU=2, EPI_ATTN=3, EPI_MULU=4, EPI_BIASF32=5, EPI_ADDX=6 };

__device__ __forceinline__ v16bf pack16(v8bf lo, v8bf hi){
    v16bf r;
#pragma unroll
    for (int i = 0; i < 8; ++i) { r[i] = lo[i]; r[i+8] = hi[i]; }
    return r;
}

// Async 16B global -> LDS copy (per lane). lds_off = LDS byte offset,
// goff = 32-bit byte offset from sgpr base.  Tracked by ASYNCcnt.
__device__ __forceinline__ void async_g2l_b128(unsigned lds_off, unsigned goff, const void* base){
    asm volatile("global_load_async_to_lds_b128 %0, %1, %2"
                 :: "v"(lds_off), "v"(goff), "s"(base) : "memory");
}
__device__ __forceinline__ void wait_async0(){
    asm volatile("s_wait_asynccnt 0" ::: "memory");
}

// C = A(bf16, MxK row-major) * B(bf16) + epilogue.
// BT=true : B given as N x K row-major rows (pre-transposed weights / k-rows) -> async path.
// BT=false: B is K x N row-major (v tiles) -> manual LDS transpose.
// CK=true : causal cap of the K loop (attn@v).
template<int EPI, bool BT, bool CK>
__global__ void __launch_bounds__(256)
gemm_bf16_wmma(const __bf16* __restrict__ A, long lda, long sAb,
               const __bf16* __restrict__ Bm, long ldb, long sBb,
               const float*  __restrict__ bias,
               const void*   __restrict__ extra, long ldE, long sEb,
               void* __restrict__ out, long ldo, long sOb,
               void* __restrict__ out2,
               int K)
{
    __shared__ __bf16 sA[2][BMT * LDSS];
    __shared__ __bf16 sB[2][BNT * LDSS];

    const int tid  = threadIdx.x;
    const int lane = tid & 31;
    const int w    = tid >> 5;
    const int m0   = blockIdx.y * BMT;
    const int n0   = blockIdx.x * BNT;
    const int bz   = blockIdx.z;

    if (EPI == EPI_ATTN && n0 > m0) return;   // strictly-above-diagonal tiles never read

    A  += (long)bz * sAb;
    Bm += (long)bz * sBb;

    const int wm = (w & 3) * 32;      // 4 waves along M
    const int wn = (w >> 2) * 64;     // 2 waves along N
    const int lr = lane & 15;
    const int kh = (lane < 16) ? 0 : 8;

    // per-thread tile-copy coordinates (2 x 16B chunks per thread per tile)
    const int cA0 = tid, cA1 = tid + 256;

    v8f acc[2][4];
#pragma unroll
    for (int mi = 0; mi < 2; ++mi)
#pragma unroll
        for (int ni = 0; ni < 4; ++ni)
#pragma unroll
            for (int r = 0; r < 8; ++r) acc[mi][ni][r] = 0.f;

    int Kt = K / BKT;
    if (CK) Kt = (m0 + BMT) / BKT;    // attn cols > m are zero -> skip those K tiles

    auto load_tiles = [&](int buf, long k0) {
        const unsigned aOff = (unsigned)(uintptr_t)&sA[buf][0];
        const unsigned bOff = (unsigned)(uintptr_t)&sB[buf][0];
        // A tile: 128x32 row-major, async
#pragma unroll
        for (int i = 0; i < 2; ++i) {
            int c = (i == 0) ? cA0 : cA1;
            int r = c >> 2, kc = (c & 3) * 8;
            async_g2l_b128(aOff + (unsigned)(r * LDSS + kc) * 2,
                           (unsigned)(((long)(m0 + r) * lda + k0 + kc) * 2), A);
        }
        if (BT) {   // B rows are N x K row-major, async
#pragma unroll
            for (int i = 0; i < 2; ++i) {
                int c = (i == 0) ? cA0 : cA1;
                int r = c >> 2, kc = (c & 3) * 8;
                async_g2l_b128(bOff + (unsigned)(r * LDSS + kc) * 2,
                               (unsigned)(((long)(n0 + r) * ldb + k0 + kc) * 2), Bm);
            }
        } else {    // K x N row-major -> transpose into LDS
#pragma unroll
            for (int i = 0; i < 2; ++i) {
                int c = (i == 0) ? cA0 : cA1;
                int kr = c >> 4, nc = (c & 15) * 8;
                v8bf vv = *(const v8bf*)&Bm[(long)(k0 + kr) * ldb + n0 + nc];
#pragma unroll
                for (int e = 0; e < 8; ++e) sB[buf][(nc + e) * LDSS + kr] = vv[e];
            }
        }
    };

    load_tiles(0, 0);                 // prologue: tile 0 -> buffer 0

    for (int kt = 0; kt < Kt; ++kt) {
        const int cur = kt & 1;
        wait_async0();                // our async writes for tile kt have landed
        __syncthreads();              // everyone's writes visible / readers of other buf done
        if (kt + 1 < Kt) load_tiles(cur ^ 1, (long)(kt + 1) * BKT);

        // fragments per ISA 16-bit A layout: lanes<16 K{kh..kh+7, kh+16..kh+23}
        v16bf af[2], bfr[4];
#pragma unroll
        for (int mi = 0; mi < 2; ++mi) {
            int r = wm + mi * 16 + lr;
            af[mi] = pack16(*(const v8bf*)&sA[cur][r * LDSS + kh],
                            *(const v8bf*)&sA[cur][r * LDSS + kh + 16]);
        }
#pragma unroll
        for (int ni = 0; ni < 4; ++ni) {
            int r = wn + ni * 16 + lr;
            bfr[ni] = pack16(*(const v8bf*)&sB[cur][r * LDSS + kh],
                             *(const v8bf*)&sB[cur][r * LDSS + kh + 16]);
        }
#pragma unroll
        for (int mi = 0; mi < 2; ++mi)
#pragma unroll
            for (int ni = 0; ni < 4; ++ni)
                acc[mi][ni] = __builtin_amdgcn_wmma_f32_16x16x32_bf16(
                    false, af[mi], false, bfr[ni], (short)0, acc[mi][ni], false, false);
    }

    // epilogue: lane n = lr, VGPR r -> row +r (lanes<16) / +8+r
    constexpr bool HAS_BIAS = (EPI != EPI_ATTN) && (EPI != EPI_MULU);
#pragma unroll
    for (int ni = 0; ni < 4; ++ni) {
        const int n = n0 + wn + ni * 16 + lr;
        const float bv = HAS_BIAS ? bias[n] : 0.f;
#pragma unroll
        for (int mi = 0; mi < 2; ++mi) {
            const int mb = m0 + wm + mi * 16 + ((lane < 16) ? 0 : 8);
#pragma unroll
            for (int r = 0; r < 8; ++r) {
                const int m = mb + r;
                const float v = acc[mi][ni][r] + bv;
                const long oi = (long)m * ldo + n;
                if (EPI == EPI_ELU) {
                    ((__bf16*)out)[oi] = (__bf16)(v > 0.f ? v : (__expf(v) - 1.f));
                } else if (EPI == EPI_DUAL) {
                    ((float*)out)[oi]   = v;
                    ((__bf16*)out2)[oi] = (__bf16)v;
                } else if (EPI == EPI_SILU) {
                    ((__bf16*)out)[oi] = (__bf16)(v / (1.f + __expf(-v)));
                } else if (EPI == EPI_ATTN) {
                    float s = v * 0.08838834764831845f;      // 1/sqrt(128)
                    float p = (n <= m && s > 0.f) ? s * s * (1.0f / 2048.0f) : 0.f;
                    ((__bf16*)out + (long)bz * sOb)[oi] = (__bf16)p;
                } else if (EPI == EPI_MULU) {
                    const __bf16* U = (const __bf16*)extra + (long)bz * sEb;
                    float uu = (float)U[(long)m * ldE + n];
                    ((__bf16*)out + (long)bz * sOb)[oi] = (__bf16)(v * uu);
                } else if (EPI == EPI_BIASF32) {
                    ((float*)out)[oi] = v;
                } else if (EPI == EPI_ADDX) {
                    const float* X = (const float*)extra;
                    ((float*)out)[oi] = v + X[(long)m * ldE + n];
                }
            }
        }
    }
}

__global__ void f32_to_bf16_k(const float* __restrict__ in, __bf16* __restrict__ out, long n)
{
    long i = (long)blockIdx.x * blockDim.x + threadIdx.x;
    long st = (long)gridDim.x * blockDim.x;
    for (; i < n; i += st) out[i] = (__bf16)in[i];
}

// out[C x R] = transpose(in[R x C]) with fp32 -> bf16, 32x32 LDS tiles.
__global__ void __launch_bounds__(256)
f32_to_bf16_tr(const float* __restrict__ in, __bf16* __restrict__ out, int R, int C)
{
    __shared__ __bf16 t[32][33];
    const int bx = blockIdx.x * 32;     // C offset
    const int by = blockIdx.y * 32;     // R offset
    const int tx = threadIdx.x & 31, ty = threadIdx.x >> 5;   // ty in 0..7
#pragma unroll
    for (int i = 0; i < 4; ++i)
        t[ty + i * 8][tx] = (__bf16)in[(long)(by + ty + i * 8) * C + bx + tx];
    __syncthreads();
#pragma unroll
    for (int i = 0; i < 4; ++i)
        out[(long)(bx + ty + i * 8) * R + by + tx] = t[tx][ty + i * 8];
}

// q = silu_qk * gamma_q + beta_q ; k = silu_qk * gamma_k + beta_k  (qk slice of uvqk)
__global__ void qk_kernel(const __bf16* __restrict__ uvqk,
                          const float* __restrict__ gq, const float* __restrict__ bq,
                          const float* __restrict__ gk, const float* __restrict__ bk,
                          __bf16* __restrict__ q, __bf16* __restrict__ k, long rows)
{
    long i = (long)blockIdx.x * blockDim.x + threadIdx.x;
    if (i >= rows * 128) return;
    long r = i >> 7; int c = (int)(i & 127);
    float v = (float)uvqk[r * 4224 + 4096 + c];
    q[i] = (__bf16)(v * gq[c] + bq[c]);
    k[i] = (__bf16)(v * gk[c] + bk[c]);
}

__device__ __forceinline__ float block_sum(float v, float* red)
{
    int t = threadIdx.x;
    red[t] = v; __syncthreads();
    for (int s = 128; s > 0; s >>= 1) {
        if (t < s) red[t] += red[t + s];
        __syncthreads();
    }
    float r = red[0]; __syncthreads();
    return r;
}

// out = LN( LN(z1; g1,b1,1e-12) + y ; g2,b2,1e-3 ), one 1024-wide row per block
__global__ void __launch_bounds__(256)
ln_final(const float* __restrict__ z1, const float* __restrict__ y,
         const float* __restrict__ g1, const float* __restrict__ b1,
         const float* __restrict__ g2, const float* __restrict__ b2,
         float* __restrict__ out)
{
    __shared__ float red[256];
    const long row = blockIdx.x;
    const int t = threadIdx.x;
    const float* zr = z1 + row * 1024;
    const float* yr = y  + row * 1024;

    float v[4]; float s = 0.f;
#pragma unroll
    for (int i = 0; i < 4; ++i) { v[i] = zr[t + i * 256]; s += v[i]; }
    float mu = block_sum(s, red) * (1.f / 1024.f);
    float s2 = 0.f;
#pragma unroll
    for (int i = 0; i < 4; ++i) { float d = v[i] - mu; s2 += d * d; }
    float inv = rsqrtf(block_sum(s2, red) * (1.f / 1024.f) + 1e-12f);

    float x2[4]; s = 0.f;
#pragma unroll
    for (int i = 0; i < 4; ++i) {
        int c = t + i * 256;
        x2[i] = (v[i] - mu) * inv * g1[c] + b1[c] + yr[c];
        s += x2[i];
    }
    float mu2 = block_sum(s, red) * (1.f / 1024.f);
    s2 = 0.f;
#pragma unroll
    for (int i = 0; i < 4; ++i) { float d = x2[i] - mu2; s2 += d * d; }
    float inv2 = rsqrtf(block_sum(s2, red) * (1.f / 1024.f) + 0.001f);
#pragma unroll
    for (int i = 0; i < 4; ++i) {
        int c = t + i * 256;
        out[row * 1024 + c] = (x2[i] - mu2) * inv2 * g2[c] + b2[c];
    }
}

extern "C" void kernel_launch(void* const* d_in, const int* in_sizes, int n_in,
                              void* d_out, int out_size, void* d_ws, size_t ws_size,
                              hipStream_t stream)
{
    (void)in_sizes; (void)n_in; (void)out_size; (void)ws_size;
    const int M = 8 * 2048;      // B*L
    const int IN = 512, H = 1024, E = 2048, SQK = 128, UV = 2 * E + SQK, L = 2048, NB = 8;

    const float* a      = (const float*)d_in[0];
    const float* Wa     = (const float*)d_in[1];
    const float* ba     = (const float*)d_in[2];
    const float* Wi     = (const float*)d_in[3];
    const float* bi     = (const float*)d_in[4];
    const float* Wuvqk  = (const float*)d_in[5];
    const float* buvqk  = (const float*)d_in[6];
    const float* gq     = (const float*)d_in[7];
    const float* bq     = (const float*)d_in[8];
    const float* gk     = (const float*)d_in[9];
    const float* bk     = (const float*)d_in[10];
    const float* Wo     = (const float*)d_in[11];
    const float* bo     = (const float*)d_in[12];
    const float* ln_gau_g = (const float*)d_in[13];
    const float* ln_gau_b = (const float*)d_in[14];
    const float* Wout   = (const float*)d_in[15];
    const float* bout   = (const float*)d_in[16];
    const float* ln_g   = (const float*)d_in[17];
    const float* ln_b   = (const float*)d_in[18];

    size_t off = 0;
    auto alloc = [&](size_t bytes) -> void* {
        void* p = (char*)d_ws + off;
        off = (off + bytes + 255) & ~(size_t)255;
        return p;
    };
    __bf16* aB     = (__bf16*)alloc((size_t)M * IN * 2);
    __bf16* WaT    = (__bf16*)alloc((size_t)IN * H * 2);   // H x IN
    __bf16* WiT    = (__bf16*)alloc((size_t)H * H * 2);    // H x H
    __bf16* WuT    = (__bf16*)alloc((size_t)H * UV * 2);   // UV x H
    __bf16* WoT    = (__bf16*)alloc((size_t)E * H * 2);    // H x E
    __bf16* WoutT  = (__bf16*)alloc((size_t)IN * H * 2);   // H x IN
    __bf16* x1B    = (__bf16*)alloc((size_t)M * H * 2);
    float*  xF     = (float*) alloc((size_t)M * H * 4);
    __bf16* xB     = (__bf16*)alloc((size_t)M * H * 2);
    __bf16* uvqkB  = (__bf16*)alloc((size_t)M * UV * 2);
    __bf16* qB     = (__bf16*)alloc((size_t)M * SQK * 2);
    __bf16* kB     = (__bf16*)alloc((size_t)M * SQK * 2);
    __bf16* attnB  = (__bf16*)alloc((size_t)NB * L * L * 2);
    __bf16* oB     = (__bf16*)alloc((size_t)M * E * 2);
    float*  z1F    = (float*) alloc((size_t)M * H * 4);
    float*  yF     = (float*) alloc((size_t)M * H * 4);

    // fp32 -> bf16 staging; weights transposed so every weight GEMM is BT (async path)
    f32_to_bf16_k<<<2048, 256, 0, stream>>>(a, aB, (long)M * IN);
    f32_to_bf16_tr<<<dim3(H  / 32, IN / 32), 256, 0, stream>>>(Wa,    WaT,   IN, H);
    f32_to_bf16_tr<<<dim3(H  / 32, H  / 32), 256, 0, stream>>>(Wi,    WiT,   H,  H);
    f32_to_bf16_tr<<<dim3(UV / 32, H  / 32), 256, 0, stream>>>(Wuvqk, WuT,   H,  UV);
    f32_to_bf16_tr<<<dim3(H  / 32, E  / 32), 256, 0, stream>>>(Wo,    WoT,   E,  H);
    f32_to_bf16_tr<<<dim3(H  / 32, IN / 32), 256, 0, stream>>>(Wout,  WoutT, IN, H);

    // x1 = elu(a @ Wa + ba)
    gemm_bf16_wmma<EPI_ELU, true, false><<<dim3(H/BNT, M/BMT, 1), 256, 0, stream>>>(
        aB, IN, 0, WaT, IN, 0, ba, nullptr, 0, 0, x1B, H, 0, nullptr, IN);
    // x = x1 @ Wi + bi   (fp32 for residual/LN + bf16 for next GEMM)
    gemm_bf16_wmma<EPI_DUAL, true, false><<<dim3(H/BNT, M/BMT, 1), 256, 0, stream>>>(
        x1B, H, 0, WiT, H, 0, bi, nullptr, 0, 0, xF, H, 0, xB, H);
    // uvqk = silu(x @ Wuvqk + b)
    gemm_bf16_wmma<EPI_SILU, true, false><<<dim3(UV/BNT, M/BMT, 1), 256, 0, stream>>>(
        xB, H, 0, WuT, H, 0, buvqk, nullptr, 0, 0, uvqkB, UV, 0, nullptr, H);
    // q,k from qk slice
    qk_kernel<<<((long)M * SQK + 255) / 256, 256, 0, stream>>>(
        uvqkB, gq, bq, gk, bk, qB, kB, M);
    // attn = causal(relu(q@k^T / sqrt(128))^2) / L   (per batch)
    gemm_bf16_wmma<EPI_ATTN, true, false><<<dim3(L/BNT, L/BMT, NB), 256, 0, stream>>>(
        qB, SQK, (long)L * SQK, kB, SQK, (long)L * SQK, nullptr,
        nullptr, 0, 0, attnB, L, (long)L * L, nullptr, SQK);
    // o = u * (attn @ v)   (per batch, causal-capped K loop; v needs LDS transpose)
    gemm_bf16_wmma<EPI_MULU, false, true><<<dim3(E/BNT, L/BMT, NB), 256, 0, stream>>>(
        attnB, L, (long)L * L, uvqkB + E, UV, (long)L * UV, nullptr,
        uvqkB, UV, (long)L * UV, oB, E, (long)L * E, nullptr, L);
    // z1 = x + (o @ Wo + bo)
    gemm_bf16_wmma<EPI_ADDX, true, false><<<dim3(H/BNT, M/BMT, 1), 256, 0, stream>>>(
        oB, E, 0, WoT, E, 0, bo, xF, H, 0, z1F, H, 0, nullptr, E);
    // y = a @ Wout + bout
    gemm_bf16_wmma<EPI_BIASF32, true, false><<<dim3(H/BNT, M/BMT, 1), 256, 0, stream>>>(
        aB, IN, 0, WoutT, IN, 0, bout, nullptr, 0, 0, yF, H, 0, nullptr, IN);
    // out = LN( LN(z1) + y )
    ln_final<<<M, 256, 0, stream>>>(z1F, yF, ln_gau_g, ln_gau_b, ln_g, ln_b, (float*)d_out);
}
